// MultiHeadAttention_18863496364738
// MI455X (gfx1250) — matrix operand order repaired and checked by reading, not compiled, not run on hardware
//
#include <hip/hip_runtime.h>
#include <hip/hip_bf16.h>

#define NHEADC 8
#define DHEADC 64
#define DMODELC 512
#define BATCHC 2
#define NQC 4096
#define NKC 4096

typedef __attribute__((ext_vector_type(16))) _Float16 v16h;
typedef __attribute__((ext_vector_type(8)))  _Float16 v8h;
typedef __attribute__((ext_vector_type(4)))  _Float16 v4h;
typedef __attribute__((ext_vector_type(8)))  float    v8f;
typedef __attribute__((ext_vector_type(4)))  float    v4f;

#if defined(__gfx1250__) && __has_builtin(__builtin_amdgcn_tensor_load_to_lds)
#define USE_TDM 1
#else
#define USE_TDM 0
#endif

#if USE_TDM
typedef unsigned int u32x4_t __attribute__((ext_vector_type(4)));
typedef int          i32x8_t __attribute__((ext_vector_type(8)));
typedef int          i32x4_t __attribute__((ext_vector_type(4)));

// TDM: DMA a 2D tile of 64 halves x 128 rows (row stride 512 halves) from
// global into LDS, inserting 4 DWORDs of pad after every 32 DWORDs (one row)
// so the LDS pitch is 72 halves. Descriptor per CDNA5 ISA 8.3-8.5.
// This toolchain exposes the 6-arg builtin:
//   (u32x4 g0, i32x8 g1, i32x4, i32x4, i32x8, i32 cpol)
__device__ __forceinline__ void tdm_load_k_tile(const _Float16* gsrc, void* ldst) {
  unsigned long long ga = (unsigned long long)(size_t)gsrc;
  u32x4_t g0;
  g0[0] = 1u;                                   // count=1, user descriptor
  g0[1] = (unsigned int)(size_t)ldst;           // lds_addr (bytes)
  g0[2] = (unsigned int)ga;                     // global_addr[31:0]
  g0[3] = (unsigned int)((ga >> 32) & 0x1FFFFFFu) | (2u << 30);  // type=2
  i32x8_t g1;
  g1[0] = (1 << 16) | (1 << 20) | (4 << 22) | (3 << 25);
  //        data_size=2B  pad_en  interval=32dw  amount=4dw
  g1[1] = (64 << 16);        // tensor_dim0 = 64   (bits 79:48)
  g1[2] = (128 << 16);       // tensor_dim1 = 128  (bits 111:80)
  g1[3] = (64 << 16);        // tile_dim0 = 64     (bits 127:112)
  g1[4] = 128;               // tile_dim1 = 128    (bits 143:128)
  g1[5] = 512;               // tensor_dim0_stride = 512 (bits 207:160)
  g1[6] = 0;
  g1[7] = 0;
  i32x4_t z4 = {0, 0, 0, 0};
  i32x8_t z8 = {0, 0, 0, 0, 0, 0, 0, 0};
  __builtin_amdgcn_tensor_load_to_lds(g0, g1, z4, z4, z8, 0);
}
#endif

// ---------------------------------------------------------------------------
// WMMA fragment loaders from LDS tiles (row-major, pitch `ldl` halves).
// Layouts per CDNA5 ISA 7.12.2 (16-bit A 16x32, 16-bit B 32x16), wave32.
// ---------------------------------------------------------------------------
__device__ __forceinline__ v16h load_frag_a(const _Float16* tile, int ldl) {
  const int lane = threadIdx.x & 31;
  const int m = lane & 15, hi = lane >> 4;
  const _Float16* p = tile + m * ldl;
  v8h lo = *(const v8h*)(p + hi * 8);
  v8h hv = *(const v8h*)(p + 16 + hi * 8);
  v16h f;
#pragma unroll
  for (int i = 0; i < 8; ++i) { f[i] = lo[i]; f[i + 8] = hv[i]; }
  return f;
}

__device__ __forceinline__ v16h load_frag_b(const _Float16* tile, int ldl) {
  const int lane = threadIdx.x & 31;
  const int n = lane & 15, hi = lane >> 4;
  const _Float16* p = tile + n * ldl;
  v8h lo = *(const v8h*)(p + hi * 8);
  v8h hv = *(const v8h*)(p + 16 + hi * 8);
  v16h f;
#pragma unroll
  for (int i = 0; i < 8; ++i) { f[i] = lo[i]; f[i + 8] = hv[i]; }
  return f;
}

// ---------------------------------------------------------------------------
// f32 -> f16 conversion (vectorized by 4)
// ---------------------------------------------------------------------------
__global__ __launch_bounds__(256) void cvt_f32_f16(const float* __restrict__ in,
                                                   _Float16* __restrict__ out,
                                                   int n4) {
  int i = blockIdx.x * blockDim.x + threadIdx.x;
  if (i < n4) {
    v4f x = ((const v4f*)in)[i];
    v4h y;
#pragma unroll
    for (int j = 0; j < 4; ++j) y[j] = (_Float16)x[j];
    ((v4h*)out)[i] = y;
  }
}

// ---------------------------------------------------------------------------
// Pack boolean mask bytes into bits: word w bit i <=> mask[w*32 + i] != 0
// ---------------------------------------------------------------------------
__global__ __launch_bounds__(256) void pack_mask(const unsigned char* __restrict__ m,
                                                 unsigned int* __restrict__ bits,
                                                 int nwords) {
  int i = blockIdx.x * blockDim.x + threadIdx.x;
  if (i < nwords) {
    const uint4* p4 = (const uint4*)(m + (long long)i * 32);
    uint4 a = p4[0], b = p4[1];
    unsigned int words[8] = {a.x, a.y, a.z, a.w, b.x, b.y, b.z, b.w};
    unsigned int w = 0;
#pragma unroll
    for (int j = 0; j < 8; ++j)
#pragma unroll
      for (int c = 0; c < 4; ++c)
        if ((words[j] >> (8 * c)) & 0xffu) w |= 1u << (j * 4 + c);
    bits[i] = w;
  }
}

// ---------------------------------------------------------------------------
// Generic C = scale * (A @ B^T) (+ bias) via WMMA f16->f32.
// Block tile 128x128, 8 waves each compute 16 rows x 128 cols.
// ---------------------------------------------------------------------------
template <bool OUT_F16, bool HAS_BIAS>
__global__ __launch_bounds__(256) void gemm_abt(
    const _Float16* __restrict__ A, const _Float16* __restrict__ Bt,
    void* __restrict__ Cv, const float* __restrict__ bias,
    int K, int lda, int ldb, int ldc, float scale) {
  __shared__ _Float16 Asub[128][40];
  __shared__ _Float16 Bsub[128][40];

  _Float16* C16 = (_Float16*)Cv;
  float*    C32 = (float*)Cv;

  const int m0 = blockIdx.y * 128;
  const int n0 = blockIdx.x * 128;
  const int t = threadIdx.x;
  const int wave = t >> 5;

  v8f acc[8];
#pragma unroll
  for (int i = 0; i < 8; ++i)
#pragma unroll
    for (int j = 0; j < 8; ++j) acc[i][j] = 0.f;

  const int lr = t >> 1;
  const int lc = (t & 1) * 16;

  for (int k0 = 0; k0 < K; k0 += 32) {
    const _Float16* ga = A  + (long long)(m0 + lr) * lda + (k0 + lc);
    const _Float16* gb = Bt + (long long)(n0 + lr) * ldb + (k0 + lc);
    v8h a0 = *(const v8h*)ga;
    v8h a1 = *(const v8h*)(ga + 8);
    v8h b0 = *(const v8h*)gb;
    v8h b1 = *(const v8h*)(gb + 8);
    *(v8h*)&Asub[lr][lc]     = a0;
    *(v8h*)&Asub[lr][lc + 8] = a1;
    *(v8h*)&Bsub[lr][lc]     = b0;
    *(v8h*)&Bsub[lr][lc + 8] = b1;
    __syncthreads();

    v16h af = load_frag_a(&Asub[wave * 16][0], 40);
#pragma unroll
    for (int nb = 0; nb < 8; ++nb) {
      v16h bf = load_frag_b(&Bsub[nb * 16][0], 40);
      acc[nb] = __builtin_amdgcn_wmma_f32_16x16x32_f16(
          false, af, false, bf, (short)0, acc[nb], false, false);
    }
    __syncthreads();
  }

  const int lane = t & 31;
  const int cn = lane & 15, hi = lane >> 4;
#pragma unroll
  for (int nb = 0; nb < 8; ++nb) {
    const int col = n0 + nb * 16 + cn;
    const float bb = HAS_BIAS ? bias[col] : 0.f;
#pragma unroll
    for (int r = 0; r < 8; ++r) {
      const int row = m0 + wave * 16 + hi * 8 + r;
      const float v = acc[nb][r] * scale + bb;
      if (OUT_F16) C16[(long long)row * ldc + col] = (_Float16)v;
      else         C32[(long long)row * ldc + col] = v;
    }
  }
}

// ---------------------------------------------------------------------------
// Fused attention core (two-pass flash-style), per (h,b) slice, 128 q-rows
// per block, 8 waves x 16 rows. Softmax runs in the exp2 domain (Q pre-scaled
// by 0.125*log2(e)). K tiles staged by the Tensor Data Mover (double
// buffered, TENSORcnt-synced) when available; V double-buffered via VGPR
// transpose. Mask consumed as packed bits with an any-set fast path.
// attn stores are non-temporal (streaming, never re-read).
// ---------------------------------------------------------------------------
__global__ __launch_bounds__(256) void attn_fused(
    const _Float16* __restrict__ qh, const _Float16* __restrict__ kh,
    const _Float16* __restrict__ vh, const unsigned int* __restrict__ mbits,
    float* __restrict__ attn, _Float16* __restrict__ oh) {
  __shared__ _Float16 Qs[128][72];
  __shared__ _Float16 Ks[2][128][72];
  __shared__ _Float16 Vs[2][64][136];
  __shared__ _Float16 Ps[128][136];

  const int z = blockIdx.z;
  const int h = z / BATCHC, b = z % BATCHC;
  const int m0 = blockIdx.y * 128;

  const _Float16* Qb = qh + (long long)b * NQC * DMODELC + h * DHEADC;
  const _Float16* Kb = kh + (long long)b * NKC * DMODELC + h * DHEADC;
  const _Float16* Vb = vh + (long long)b * NKC * DMODELC + h * DHEADC;
  const unsigned int* MbW = mbits + (long long)b * NQC * (NKC / 32);
  float* Ab = attn + (long long)z * NQC * NKC;
  _Float16* Ob = oh + (long long)b * NQC * DMODELC + h * DHEADC;

  const int t = threadIdx.x, wave = t >> 5, lane = t & 31;
  const int cn = lane & 15, hi = lane >> 4;
  const float NINF = -__builtin_inff();

  const int lr = t >> 1, lc = (t & 1) * 32;   // Q/K tile loader indices
  const int qd = t >> 3, bn = (t & 7) * 8;    // V tile loader indices

  // Q tile: loaded once, pre-scaled by (1/sqrt(dk)) * log2(e).
  {
    const _Float16 QSC = (_Float16)0.18033688f;  // 0.125 * log2(e)
    const _Float16* g = Qb + (long long)(m0 + lr) * DMODELC + lc;
    v8h q0 = *(const v8h*)(g);
    v8h q1 = *(const v8h*)(g + 8);
    v8h q2 = *(const v8h*)(g + 16);
    v8h q3 = *(const v8h*)(g + 24);
#pragma unroll
    for (int j = 0; j < 8; ++j) {
      q0[j] *= QSC; q1[j] *= QSC; q2[j] *= QSC; q3[j] *= QSC;
    }
    *(v8h*)&Qs[lr][lc]      = q0;
    *(v8h*)&Qs[lr][lc + 8]  = q1;
    *(v8h*)&Qs[lr][lc + 16] = q2;
    *(v8h*)&Qs[lr][lc + 24] = q3;
  }

  float m_run[8], l_run[8];
#pragma unroll
  for (int r = 0; r < 8; ++r) { m_run[r] = NINF; l_run[r] = 0.f; }

  v8f oacc[4];
#pragma unroll
  for (int i = 0; i < 4; ++i)
#pragma unroll
    for (int j = 0; j < 8; ++j) oacc[i][j] = 0.f;

  const int NSTEP = NKC / 128;

#if USE_TDM
#define STAGE_K(step, buf)                                                    \
  {                                                                           \
    if (wave == 0)                                                            \
      tdm_load_k_tile(Kb + (long long)(step) * 128 * DMODELC,                 \
                      (void*)&Ks[buf][0][0]);                                 \
  }
#define WAIT_K()                                                              \
  {                                                                           \
    if (wave == 0) __builtin_amdgcn_s_wait_tensorcnt((unsigned short)0);      \
  }
#else
#define STAGE_K(step, buf)                                                    \
  {                                                                           \
    const _Float16* g = Kb + (long long)((step) * 128 + lr) * DMODELC + lc;   \
    v8h a0 = *(const v8h*)(g);                                                \
    v8h a1 = *(const v8h*)(g + 8);                                            \
    v8h a2 = *(const v8h*)(g + 16);                                           \
    v8h a3 = *(const v8h*)(g + 24);                                           \
    *(v8h*)&Ks[buf][lr][lc]      = a0;                                        \
    *(v8h*)&Ks[buf][lr][lc + 8]  = a1;                                        \
    *(v8h*)&Ks[buf][lr][lc + 16] = a2;                                        \
    *(v8h*)&Ks[buf][lr][lc + 24] = a3;                                        \
  }
#define WAIT_K()
#endif

#define STAGE_V(step, buf)                                                    \
  {                                                                           \
    const _Float16* gv = Vb + (long long)((step) * 128 + 4 * qd) * DMODELC + bn; \
    v8h r0 = *(const v8h*)(gv);                                               \
    v8h r1 = *(const v8h*)(gv + DMODELC);                                     \
    v8h r2 = *(const v8h*)(gv + 2 * DMODELC);                                 \
    v8h r3 = *(const v8h*)(gv + 3 * DMODELC);                                 \
    _Pragma("unroll") for (int i = 0; i < 8; ++i) {                           \
      v4h pk;                                                                 \
      pk[0] = r0[i]; pk[1] = r1[i]; pk[2] = r2[i]; pk[3] = r3[i];             \
      *(v4h*)&Vs[buf][bn + i][4 * qd] = pk;                                   \
    }                                                                         \
  }

  // ------------------------- Pass 1: statistics --------------------------
  STAGE_K(0, 0);
  for (int i = 0; i < NSTEP; ++i) {
    WAIT_K();
    __syncthreads();  // buffer i&1 visible to all; prev reads complete
    if (i + 1 < NSTEP) STAGE_K(i + 1, (i + 1) & 1);
    const int k0 = i * 128;

    v8f sacc[8];
#pragma unroll
    for (int a = 0; a < 8; ++a)
#pragma unroll
      for (int j = 0; j < 8; ++j) sacc[a][j] = 0.f;

#pragma unroll
    for (int ks = 0; ks < 64; ks += 32) {
      v16h af = load_frag_a(&Qs[wave * 16][ks], 72);
#pragma unroll
      for (int nb = 0; nb < 8; ++nb) {
        v16h bf = load_frag_b(&Ks[i & 1][nb * 16][ks], 72);
        sacc[nb] = __builtin_amdgcn_wmma_f32_16x16x32_f16(
            false, af, false, bf, (short)0, sacc[nb], false, false);
      }
    }

    // online softmax statistics (exp2 domain), mask any-set fast path
#pragma unroll
    for (int r = 0; r < 8; ++r) {
      const long long row = m0 + wave * 16 + hi * 8 + r;
      const uint4 mq = *(const uint4*)(MbW + row * (NKC / 32) + (k0 >> 5));
      const unsigned int mw[4] = {mq.x, mq.y, mq.z, mq.w};
      float mx = NINF;
      if ((mw[0] | mw[1] | mw[2] | mw[3]) == 0u) {
#pragma unroll
        for (int nb = 0; nb < 8; ++nb) mx = fmaxf(mx, sacc[nb][r]);
      } else {
#pragma unroll
        for (int nb = 0; nb < 8; ++nb) {
          float s = sacc[nb][r];
          if ((mw[nb >> 1] >> ((nb & 1) * 16 + cn)) & 1u) s = NINF;
          sacc[nb][r] = s;
          mx = fmaxf(mx, s);
        }
      }
#pragma unroll
      for (int d = 1; d < 16; d <<= 1) mx = fmaxf(mx, __shfl_xor(mx, d, 32));
      const float mnew = fmaxf(m_run[r], mx);
      float ssum = 0.f;
#pragma unroll
      for (int nb = 0; nb < 8; ++nb)
        ssum += __builtin_amdgcn_exp2f(sacc[nb][r] - mnew);
#pragma unroll
      for (int d = 1; d < 16; d <<= 1) ssum += __shfl_xor(ssum, d, 32);
      l_run[r] = l_run[r] * __builtin_amdgcn_exp2f(m_run[r] - mnew) + ssum;
      m_run[r] = mnew;
    }
  }

  float inv_l[8];
#pragma unroll
  for (int r = 0; r < 8; ++r) inv_l[r] = 1.f / l_run[r];

  // --------------------- Pass 2: emit attn + O accum ---------------------
  STAGE_K(0, 0);
  STAGE_V(0, 0);
  for (int i = 0; i < NSTEP; ++i) {
    WAIT_K();
    __syncthreads();
    if (i + 1 < NSTEP) {
      STAGE_K(i + 1, (i + 1) & 1);
      STAGE_V(i + 1, (i + 1) & 1);
    }
    const int k0 = i * 128;

    v8f sacc[8];
#pragma unroll
    for (int a = 0; a < 8; ++a)
#pragma unroll
      for (int j = 0; j < 8; ++j) sacc[a][j] = 0.f;

#pragma unroll
    for (int ks = 0; ks < 64; ks += 32) {
      v16h af = load_frag_a(&Qs[wave * 16][ks], 72);
#pragma unroll
      for (int nb = 0; nb < 8; ++nb) {
        v16h bf = load_frag_b(&Ks[i & 1][nb * 16][ks], 72);
        sacc[nb] = __builtin_amdgcn_wmma_f32_16x16x32_f16(
            false, af, false, bf, (short)0, sacc[nb], false, false);
      }
    }

    // p = exp2(s - m); stream normalized attn; stage e (<=1) as f16 in LDS.
#pragma unroll
    for (int r = 0; r < 8; ++r) {
      const long long row = m0 + wave * 16 + hi * 8 + r;
      const uint4 mq = *(const uint4*)(MbW + row * (NKC / 32) + (k0 >> 5));
      const unsigned int mw[4] = {mq.x, mq.y, mq.z, mq.w};
      const float mr = m_run[r], il = inv_l[r];
      if ((mw[0] | mw[1] | mw[2] | mw[3]) == 0u) {
#pragma unroll
        for (int nb = 0; nb < 8; ++nb) {
          const float e = __builtin_amdgcn_exp2f(sacc[nb][r] - mr);
          __builtin_nontemporal_store(e * il,
                                      &Ab[row * NKC + (k0 + nb * 16 + cn)]);
          Ps[wave * 16 + hi * 8 + r][nb * 16 + cn] = (_Float16)e;
        }
      } else {
#pragma unroll
        for (int nb = 0; nb < 8; ++nb) {
          float s = sacc[nb][r];
          if ((mw[nb >> 1] >> ((nb & 1) * 16 + cn)) & 1u) s = NINF;
          const float e = __builtin_amdgcn_exp2f(s - mr);
          __builtin_nontemporal_store(e * il,
                                      &Ab[row * NKC + (k0 + nb * 16 + cn)]);
          Ps[wave * 16 + hi * 8 + r][nb * 16 + cn] = (_Float16)e;
        }
      }
    }

    // O += P(16x128) @ V(128x64); A-frags read only this wave's own rows,
    // so in-wave DScnt ordering suffices (no extra barrier needed).
#pragma unroll
    for (int ks2 = 0; ks2 < 128; ks2 += 32) {
      v16h af = load_frag_a(&Ps[wave * 16][ks2], 136);
#pragma unroll
      for (int nb2 = 0; nb2 < 4; ++nb2) {
        v16h bf = load_frag_b(&Vs[i & 1][nb2 * 16][ks2], 136);
        oacc[nb2] = __builtin_amdgcn_wmma_f32_16x16x32_f16(
            false, af, false, bf, (short)0, oacc[nb2], false, false);
      }
    }
  }
#undef STAGE_K
#undef STAGE_V
#undef WAIT_K

  // epilogue: fold 1/l into O
#pragma unroll
  for (int nb2 = 0; nb2 < 4; ++nb2) {
    const int col = nb2 * 16 + cn;
#pragma unroll
    for (int r = 0; r < 8; ++r) {
      const long long row = m0 + wave * 16 + hi * 8 + r;
      Ob[row * DMODELC + col] = (_Float16)(oacc[nb2][r] * inv_l[r]);
    }
  }
}

// ---------------------------------------------------------------------------
// Host-side orchestration
// ---------------------------------------------------------------------------
extern "C" void kernel_launch(void* const* d_in, const int* in_sizes, int n_in,
                              void* d_out, int out_size, void* d_ws, size_t ws_size,
                              hipStream_t stream) {
  (void)in_sizes; (void)n_in; (void)out_size; (void)ws_size;
  const float* q    = (const float*)d_in[0];
  const float* k    = (const float*)d_in[1];
  const float* v    = (const float*)d_in[2];
  const unsigned char* mask = (const unsigned char*)d_in[3];
  const float* Wq   = (const float*)d_in[4];
  const float* bq   = (const float*)d_in[5];
  const float* Wk   = (const float*)d_in[6];
  const float* bk   = (const float*)d_in[7];
  const float* Wv   = (const float*)d_in[8];
  const float* bv   = (const float*)d_in[9];
  const float* Wo   = (const float*)d_in[10];
  const float* bo   = (const float*)d_in[11];

  float* attn_out = (float*)d_out;                                     // [H*B, NQ, NK]
  float* proj_out = attn_out + (long long)NHEADC * BATCHC * NQC * NKC; // [B, NQ, DMODEL]

  _Float16* ws = (_Float16*)d_ws;
  const long long SZ_X = (long long)BATCHC * NQC * DMODELC;
  const long long SZ_W = (long long)DMODELC * DMODELC;
  _Float16* q16  = ws;
  _Float16* k16  = q16 + SZ_X;
  _Float16* v16  = k16 + SZ_X;
  _Float16* wq16 = v16 + SZ_X;
  _Float16* wk16 = wq16 + SZ_W;
  _Float16* wv16 = wk16 + SZ_W;
  _Float16* wo16 = wv16 + SZ_W;
  _Float16* qh16 = wo16 + SZ_W;
  _Float16* kh16 = qh16 + SZ_X;
  _Float16* vh16 = kh16 + SZ_X;
  _Float16* oh16 = vh16 + SZ_X;
  unsigned int* mbitsw = (unsigned int*)(oh16 + SZ_X);   // 16B-aligned

  const dim3 blk(256);

  // 1) convert inputs + weights to f16; pack mask to bits
  {
    int n4x = (int)(SZ_X / 4), n4w = (int)(SZ_W / 4);
    cvt_f32_f16<<<dim3((n4x + 255) / 256), blk, 0, stream>>>(q, q16, n4x);
    cvt_f32_f16<<<dim3((n4x + 255) / 256), blk, 0, stream>>>(k, k16, n4x);
    cvt_f32_f16<<<dim3((n4x + 255) / 256), blk, 0, stream>>>(v, v16, n4x);
    cvt_f32_f16<<<dim3((n4w + 255) / 256), blk, 0, stream>>>(Wq, wq16, n4w);
    cvt_f32_f16<<<dim3((n4w + 255) / 256), blk, 0, stream>>>(Wk, wk16, n4w);
    cvt_f32_f16<<<dim3((n4w + 255) / 256), blk, 0, stream>>>(Wv, wv16, n4w);
    cvt_f32_f16<<<dim3((n4w + 255) / 256), blk, 0, stream>>>(Wo, wo16, n4w);
    int nwords = (int)((long long)BATCHC * NQC * NKC / 32);
    pack_mask<<<dim3((nwords + 255) / 256), blk, 0, stream>>>(mask, mbitsw, nwords);
  }

  // 2) QKV projections
  {
    dim3 grid(DMODELC / 128, (BATCHC * NQC) / 128, 1);
    gemm_abt<true, true><<<grid, blk, 0, stream>>>(
        q16, wq16, (void*)qh16, bq, DMODELC, DMODELC, DMODELC, DMODELC, 1.0f);
    gemm_abt<true, true><<<grid, blk, 0, stream>>>(
        k16, wk16, (void*)kh16, bk, DMODELC, DMODELC, DMODELC, DMODELC, 1.0f);
    gemm_abt<true, true><<<grid, blk, 0, stream>>>(
        v16, wv16, (void*)vh16, bv, DMODELC, DMODELC, DMODELC, DMODELC, 1.0f);
  }

  // 3) fused scores + softmax + attn emit + attn@V
  {
    dim3 grid(1, NQC / 128, NHEADC * BATCHC);
    attn_fused<<<grid, blk, 0, stream>>>(qh16, kh16, vh16, mbitsw,
                                         attn_out, oh16);
  }

  // 4) output = out_heads @ Wo^T + bo
  {
    dim3 grid(DMODELC / 128, (BATCHC * NQC) / 128, 1);
    gemm_abt<false, true><<<grid, blk, 0, stream>>>(
        oh16, wo16, (void*)proj_out, bo, DMODELC, DMODELC, DMODELC, DMODELC,
        1.0f);
  }
}
